// DIIN_62070867362599
// MI455X (gfx1250) — matrix operand order repaired and checked by reading, not compiled, not run on hardware
//
#include <hip/hip_runtime.h>
#include <hip/hip_bf16.h>
#include <cstdint>

// ---------------------------------------------------------------------------
// DIIN pipeline for MI455X (gfx1250). All GEMM-shaped work through
// v_wmma_f32_16x16x32_bf16 (wave32), fp32 accumulation, bf16 storage.
// All WMMA operands are laid out so every fragment is two 16B loads:
//  - A operands stored in natural k order (shuf32 is the identity map).
//  - B operands stored transposed (N x K row-major) = "Bt".
//  - CNN trunk in NHWC with channel segments padded to 32.
// ---------------------------------------------------------------------------

typedef __attribute__((ext_vector_type(16))) __bf16 v16bf;
typedef __attribute__((ext_vector_type(8)))  float  v8f;

#define DEV __device__ __forceinline__

static constexpr int BATCH = 8;
static constexpr int SEQ   = 96;    // L
static constexpr int DIM   = 300;   // D
static constexpr int DP    = 320;   // D padded to 32
static constexpr int ROWS  = BATCH * SEQ;  // 768

// Identity k-map: A is stored in natural k order, matching the Bt operands
// (each half-wave reads 16 contiguous k, same as B).
__host__ __device__ inline int shuf32(int k) {
  int r = k & 31;
  int p = r;
  return (k & ~31) | p;
}

DEV v8f wmma_bf16(v16bf a, v16bf b, v8f c) {
  return __builtin_amdgcn_wmma_f32_16x16x32_bf16(
      false, a, false, b, (short)0, c, false, false);
}

DEV v16bf load16(const __bf16* __restrict__ p) {
  union { uint4 q[2]; v16bf f; } u;
  const uint4* q = (const uint4*)p;
  u.q[0] = q[0];
  u.q[1] = q[1];
  return u.f;
}

DEV v16bf zero16() {
  union { uint4 q[2]; v16bf f; } u;
  u.q[0] = make_uint4(0u, 0u, 0u, 0u);
  u.q[1] = make_uint4(0u, 0u, 0u, 0u);
  return u.f;
}

// ---------------------------------------------------------------------------
// Generic batched GEMM: C[M,N] = A_sh[M,Kp] * Bt[N,Kp]^T (+bias over N).
// 4 waves/block (one m-tile each), NT=2 n-tiles per wave sharing the A frag.
// TRANSC=true stores C transposed (N-major, contiguous per lane) for NHWC.
// ---------------------------------------------------------------------------
template <typename OutT, bool TRANSC>
__global__ __launch_bounds__(128)
void k_gemm(const __bf16* __restrict__ A, int lda, long long strideA,
            const __bf16* __restrict__ Bt, int ldbt, long long strideB,
            OutT* __restrict__ C, int ldc, long long strideC,
            int Kp, int Mtiles,
            const float* __restrict__ bias, int biasN) {
  int wave = threadIdx.x >> 5;
  int lane = threadIdx.x & 31;
  int hi = lane >> 4, nl = lane & 15;
  int m_tile = blockIdx.y * 4 + wave;
  if (m_tile >= Mtiles) return;
  const __bf16* Ab = A + (long long)blockIdx.z * strideA;
  const __bf16* Bb = Bt + (long long)blockIdx.z * strideB;
  OutT* Cb = C + (long long)blockIdx.z * strideC;
  int n0 = blockIdx.x * 32 + nl;
  int n1 = n0 + 16;
  int mrow = m_tile * 16 + nl;
  const __bf16* ap = Ab + (size_t)mrow * lda + hi * 16;
  const __bf16* bp0 = Bb + (size_t)n0 * ldbt + hi * 16;
  const __bf16* bp1 = Bb + (size_t)n1 * ldbt + hi * 16;
  v8f acc0 = {}, acc1 = {};
  for (int k0 = 0; k0 < Kp; k0 += 32) {
    if (k0 + 32 < Kp) {
      __builtin_prefetch(ap + k0 + 32, 0, 1);   // global_prefetch_b8
      __builtin_prefetch(bp0 + k0 + 32, 0, 1);
    }
    v16bf fa = load16(ap + k0);
    v16bf f0 = load16(bp0 + k0);
    v16bf f1 = load16(bp1 + k0);
    acc0 = wmma_bf16(fa, f0, acc0);
    acc1 = wmma_bf16(fa, f1, acc1);
  }
  if (TRANSC) {
#pragma unroll
    for (int r = 0; r < 8; ++r) {
      int m = m_tile * 16 + r + hi * 8;
      Cb[(size_t)n0 * ldc + m] = (OutT)acc0[r];
      Cb[(size_t)n1 * ldc + m] = (OutT)acc1[r];
    }
  } else {
    float bv0 = (bias != nullptr && n0 < biasN) ? bias[n0] : 0.0f;
    float bv1 = (bias != nullptr && n1 < biasN) ? bias[n1] : 0.0f;
#pragma unroll
    for (int r = 0; r < 8; ++r) {
      int m = m_tile * 16 + r + hi * 8;
      Cb[(size_t)m * ldc + n0] = (OutT)(acc0[r] + bv0);
      Cb[(size_t)m * ldc + n1] = (OutT)(acc1[r] + bv1);
    }
  }
}

// ---------------------------------------------------------------------------
// Interaction + 1x1 icnn conv (NHWC output):
// fm[b,p,h,o] = sum_d icnn[o,d] * p_enc[b,p,d] * h_enc[b,h,d] + icnn_b[o]
// A built on the fly from two shuffled b128 loads; Bt = h_enc rows directly.
// ---------------------------------------------------------------------------
__global__ __launch_bounds__(128)
void k_interact(const __bf16* __restrict__ Aic,    // 96 x 320 shuffled
                const __bf16* __restrict__ p_sh,   // 768 x 320 shuffled
                const __bf16* __restrict__ h_nat,  // 768 x 320 natural
                const float* __restrict__ icnn_b,  // 90
                __bf16* __restrict__ act, int Cp) {
  int wave = threadIdx.x >> 5;
  int lane = threadIdx.x & 31;
  int hi = lane >> 4, nl = lane & 15;
  int m_tile = blockIdx.y * 4 + wave;
  if (m_tile >= 6) return;
  int bp = blockIdx.z;
  int b = bp / SEQ, prow = bp - b * SEQ;
  int n = blockIdx.x * 16 + nl;  // h position
  int mrow = m_tile * 16 + nl;
  const __bf16* wp = Aic + (size_t)mrow * DP + hi * 16;
  const __bf16* pp = p_sh + (size_t)bp * DP + hi * 16;
  const __bf16* hp = h_nat + ((size_t)b * SEQ + n) * DP + hi * 16;
  v8f acc = {};
  for (int k0 = 0; k0 < DP; k0 += 32) {
    v16bf wv = load16(wp + k0);
    v16bf pv = load16(pp + k0);
    v16bf fa;
#pragma unroll
    for (int e = 0; e < 16; ++e)
      fa[e] = (__bf16)((float)wv[e] * (float)pv[e]);
    v16bf fb = load16(hp + k0);
    acc = wmma_bf16(fa, fb, acc);
  }
  size_t obase = (((size_t)b * SEQ + prow) * SEQ + n) * Cp;
#pragma unroll
  for (int r = 0; r < 8; ++r) {
    int o = m_tile * 16 + r + hi * 8;
    if (o < 90) act[obase + o] = (__bf16)(acc[r] + icnn_b[o]);
  }
}

// ---------------------------------------------------------------------------
// Dense 3x3 conv (pad=1), NHWC, implicit GEMM with per-(kh,kw) channel
// segments padded to Cpk (multiple of 32). M tile = 32 (20 valid), fused
// ReLU, output written at channel offset Cin (dense concat in-place).
// ---------------------------------------------------------------------------
__global__ __launch_bounds__(128)
void k_dense_conv(const __bf16* __restrict__ Aw, int lda,  // 32 x 9*Cpk
                  __bf16* __restrict__ act, int Cp, int H, int W,
                  int Cin, int Cpk) {
  int wave = threadIdx.x >> 5;
  int lane = threadIdx.x & 31;
  int hi = lane >> 4, nl = lane & 15;
  int b = blockIdx.z;
  int n = (blockIdx.x * 4 + wave) * 16 + nl;
  int ph = n / W, pw = n - ph * W;
  __bf16* xb = act + (size_t)b * H * W * Cp;
  const __bf16* a0 = Aw + (size_t)nl * lda + hi * 16;
  const __bf16* a1 = Aw + (size_t)(16 + nl) * lda + hi * 16;
  int Kp = 9 * Cpk;
  v8f acc0 = {}, acc1 = {};
  for (int k0 = 0; k0 < Kp; k0 += 32) {
    int s = k0 / Cpk;          // 3x3 tap index (32-block never straddles taps)
    int c0 = k0 - s * Cpk;
    int kh = s / 3, kw = s - kh * 3;
    int y = ph + kh - 1, x = pw + kw - 1;
    v16bf fb = (y >= 0 && y < H && x >= 0 && x < W)
                   ? load16(xb + ((size_t)y * W + x) * Cp + c0 + hi * 16)
                   : zero16();
    v16bf fa0 = load16(a0 + k0);
    v16bf fa1 = load16(a1 + k0);
    acc0 = wmma_bf16(fa0, fb, acc0);
    acc1 = wmma_bf16(fa1, fb, acc1);
  }
  size_t obase = (size_t)n * Cp + Cin + (size_t)b * H * W * Cp;
#pragma unroll
  for (int r = 0; r < 8; ++r) {
    int c = hi * 8 + r;
    float v0 = acc0[r] > 0.0f ? acc0[r] : 0.0f;
    xb[(size_t)n * Cp + Cin + c] = (__bf16)v0;
    int c1 = 16 + c;
    if (c1 < 20) {
      float v1 = acc1[r] > 0.0f ? acc1[r] : 0.0f;
      xb[(size_t)n * Cp + Cin + c1] = (__bf16)v1;
    }
  }
  (void)obase;
}

// ---------------------------------------------------------------------------
// Weight conversion kernels
// ---------------------------------------------------------------------------
// A-side: pad + k-shuffle. dst[r][kp] = src[r][shuf32(kp)] (shuffle is its
// own inverse).
__global__ void k_padw_sh(const float* __restrict__ src, __bf16* __restrict__ dst,
                          int sR, int sC, int dR, int dC) {
  int idx = blockIdx.x * blockDim.x + threadIdx.x;
  if (idx >= dR * dC) return;
  int r = idx / dC, kp = idx - r * dC;
  int k = shuf32(kp);
  float v = (r < sR && k < sC) ? src[(size_t)r * sC + k] : 0.0f;
  dst[idx] = (__bf16)v;
}

// B-side: transpose K x N source into N x K "Bt" (natural k order).
__global__ void k_padw_t(const float* __restrict__ src, __bf16* __restrict__ dst,
                         int sK, int sN, int dR, int dC) {
  int idx = blockIdx.x * blockDim.x + threadIdx.x;
  if (idx >= dR * dC) return;
  int n = idx / dC, k = idx - n * dC;
  float v = (k < sK && n < sN) ? src[(size_t)k * sN + n] : 0.0f;
  dst[idx] = (__bf16)v;
}

// Conv weights: (20,Cin,3,3) -> (32, 9*Cpk), tap-major / channel-minor,
// channel segments padded to Cpk, k-shuffled.
__global__ void k_padw_conv(const float* __restrict__ src, __bf16* __restrict__ dst,
                            int Cin, int Cpk) {
  int dC = 9 * Cpk;
  int idx = blockIdx.x * blockDim.x + threadIdx.x;
  if (idx >= 32 * dC) return;
  int m = idx / dC, kp = idx - m * dC;
  int k = shuf32(kp);
  int s = k / Cpk, c = k - s * Cpk;
  int kh = s / 3, kw = s - kh * 3;
  float v = (m < 20 && c < Cin)
                ? src[(((size_t)m * Cin + c) * 3 + kh) * 3 + kw]
                : 0.0f;
  dst[idx] = (__bf16)v;
}

__global__ void k_zero(uint4* __restrict__ p, long long n) {
  long long idx = (long long)blockIdx.x * blockDim.x + threadIdx.x;
  if (idx < n) p[idx] = make_uint4(0u, 0u, 0u, 0u);
}

// ---------------------------------------------------------------------------
// Elementwise kernels (write shuffled A-copies and/or natural Bt-copies)
// ---------------------------------------------------------------------------
__global__ void k_gather(const int* __restrict__ tok, const float* __restrict__ emb,
                         __bf16* __restrict__ out_sh) {
  int idx = blockIdx.x * blockDim.x + threadIdx.x;
  if (idx >= ROWS * DP) return;
  int r = idx / DP, d = idx - r * DP;
  int t = tok[r];
  float v = (t != 0 && d < DIM) ? emb[(size_t)t * DIM + d] : 0.0f;
  out_sh[(size_t)r * DP + shuf32(d)] = (__bf16)v;
}

__global__ void k_xcvt(const float* __restrict__ src, __bf16* __restrict__ dst_sh,
                       __bf16* __restrict__ dst_nat) {
  int idx = blockIdx.x * blockDim.x + threadIdx.x;
  if (idx >= ROWS * DP) return;
  int r = idx / DP, c = idx - r * DP;
  float v = (c < DIM) ? src[idx] : 0.0f;
  dst_sh[(size_t)r * DP + shuf32(c)] = (__bf16)v;
  dst_nat[idx] = (__bf16)v;
}

__global__ void k_highway(const float* __restrict__ pg, const float* __restrict__ pn,
                          const float* __restrict__ pl, float* __restrict__ x,
                          __bf16* __restrict__ x_sh, __bf16* __restrict__ x_nat) {
  int idx = blockIdx.x * blockDim.x + threadIdx.x;
  if (idx >= ROWS * DP) return;
  int r = idx / DP, c = idx - r * DP;
  float v = 0.0f;
  if (c < DIM) {
    float g = 1.0f / (1.0f + __expf(-pg[idx]));
    v = g * tanhf(pn[idx]) + (1.0f - g) * pl[idx];
  }
  x[idx] = v;
  x_sh[(size_t)r * DP + shuf32(c)] = (__bf16)v;
  x_nat[idx] = (__bf16)v;
}

__global__ void k_attn_vec(const float* __restrict__ x, const float* __restrict__ attn_w,
                           float* __restrict__ up, float* __restrict__ mid,
                           __bf16* __restrict__ xw_sh) {
  int r = blockIdx.x;
  int tid = threadIdx.x;
  __shared__ float s_up[128], s_mid[128];
  float su = 0.0f, sm = 0.0f;
  for (int d = tid; d < DIM; d += 128) {
    float xv = x[(size_t)r * DP + d];
    su += xv * attn_w[d];
    sm += xv * attn_w[DIM + d];
    xw_sh[(size_t)r * DP + shuf32(d)] = (__bf16)(xv * attn_w[2 * DIM + d]);
  }
  for (int d = DIM + tid; d < DP; d += 128)
    xw_sh[(size_t)r * DP + shuf32(d)] = (__bf16)0.0f;
  s_up[tid] = su; s_mid[tid] = sm;
  __syncthreads();
  for (int s = 64; s > 0; s >>= 1) {
    if (tid < s) { s_up[tid] += s_up[tid + s]; s_mid[tid] += s_mid[tid + s]; }
    __syncthreads();
  }
  if (tid == 0) { up[r] = s_up[0]; mid[r] = s_mid[0]; }
}

// xT[b][d][l] (natural) from enc fp32 -- Bt operand for the att GEMM.
__global__ void k_transpose(const float* __restrict__ enc, __bf16* __restrict__ outT) {
  int idx = blockIdx.x * blockDim.x + threadIdx.x;
  if (idx >= BATCH * DP * SEQ) return;
  int l = idx % SEQ;
  int t = idx / SEQ;
  int d = t % DP;
  int b = t / DP;
  float v = (d < DIM) ? enc[((size_t)b * SEQ + l) * DP + d] : 0.0f;
  outT[idx] = (__bf16)v;
}

__global__ void k_softmax(const float* __restrict__ S, const float* __restrict__ up,
                          const float* __restrict__ mid, __bf16* __restrict__ Sbf_sh) {
  int row = blockIdx.x;  // b*96 + i
  int b = row / SEQ;
  int tid = threadIdx.x;
  __shared__ float red[128];
  const float* Srow = S + (size_t)row * SEQ;
  float myv = 0.0f, v = -1e30f;
  if (tid < SEQ) {
    myv = Srow[tid] + up[b * SEQ + tid] + mid[row];
    v = myv;
  }
  red[tid] = v;
  __syncthreads();
  for (int s = 64; s > 0; s >>= 1) {
    if (tid < s) red[tid] = fmaxf(red[tid], red[tid + s]);
    __syncthreads();
  }
  float mx = red[0];
  __syncthreads();
  float e = (tid < SEQ) ? __expf(myv - mx) : 0.0f;
  red[tid] = e;
  __syncthreads();
  for (int s = 64; s > 0; s >>= 1) {
    if (tid < s) red[tid] += red[tid + s];
    __syncthreads();
  }
  float inv = 1.0f / red[0];
  if (tid < SEQ) Sbf_sh[(size_t)row * SEQ + shuf32(tid)] = (__bf16)(e * inv);
}

__global__ void k_cat(const float* __restrict__ x, const float* __restrict__ att,
                      __bf16* __restrict__ cat_sh) {
  int idx = blockIdx.x * blockDim.x + threadIdx.x;
  if (idx >= ROWS * 640) return;
  int r = idx / 640, c = idx - r * 640;
  float v = 0.0f;
  if (c < DIM) v = x[(size_t)r * DP + c];
  else if (c < 2 * DIM) v = att[(size_t)r * DP + (c - DIM)];
  cat_sh[(size_t)r * 640 + shuf32(c)] = (__bf16)v;
}

__global__ void k_fuse(const float* __restrict__ pre, const float* __restrict__ x,
                       __bf16* __restrict__ enc_sh, __bf16* __restrict__ enc_nat) {
  int idx = blockIdx.x * blockDim.x + threadIdx.x;
  if (idx >= ROWS * DP) return;
  int r = idx / DP, c = idx - r * DP;
  float v = 0.0f;
  if (c < DIM) {
    float p = pre[idx];
    float g = 1.0f / (1.0f + __expf(-p));
    v = g * x[idx] + g * tanhf(p);
  }
  enc_sh[(size_t)r * DP + shuf32(c)] = (__bf16)v;
  enc_nat[idx] = (__bf16)v;
}

// NHWC 2x2 maxpool
__global__ void k_maxpool(const __bf16* __restrict__ in, __bf16* __restrict__ out,
                          int Cin, int Cinp, int H, int W, int Coutp, int total) {
  int idx = blockIdx.x * blockDim.x + threadIdx.x;
  if (idx >= total) return;
  int Ho = H >> 1, Wo = W >> 1;
  int c = idx % Cin; int t = idx / Cin;
  int wo = t % Wo; t /= Wo;
  int ho = t % Ho; int b = t / Ho;
  const __bf16* p0 = in + (((size_t)b * H + 2 * ho) * W + 2 * wo) * Cinp + c;
  const __bf16* p1 = p0 + (size_t)W * Cinp;
  float m0 = fmaxf((float)p0[0], (float)p0[Cinp]);
  float m1 = fmaxf((float)p1[0], (float)p1[Cinp]);
  out[(((size_t)b * Ho + ho) * Wo + wo) * Coutp + c] = (__bf16)fmaxf(m0, m1);
}

// final FC over NHWC act4: flat f = c*144 + s (NCHW flatten order)
__global__ void k_fc(const __bf16* __restrict__ act, const float* __restrict__ w,
                     const float* __restrict__ bv, float* __restrict__ out) {
  int b = blockIdx.x / 3, o = blockIdx.x % 3;
  int tid = threadIdx.x;
  __shared__ float red[256];
  float s = 0.0f;
  for (int f = tid; f < 21744; f += 256) {
    int c = f / 144;
    int si = f - c * 144;
    float a = (float)act[((size_t)b * 144 + si) * 152 + c];
    s += a * w[(size_t)f * 3 + o];
  }
  red[tid] = s;
  __syncthreads();
  for (int st = 128; st > 0; st >>= 1) {
    if (tid < st) red[tid] += red[tid + st];
    __syncthreads();
  }
  if (tid == 0) out[b * 3 + o] = red[0] + bv[o];
}

// ---------------------------------------------------------------------------
// Host orchestration
// ---------------------------------------------------------------------------
static inline int rup(int v, int a) { return (v + a - 1) & ~(a - 1); }

extern "C" void kernel_launch(void* const* d_in, const int* in_sizes, int n_in,
                              void* d_out, int out_size, void* d_ws, size_t ws_size,
                              hipStream_t stream) {
  (void)in_sizes; (void)n_in; (void)out_size; (void)ws_size;
  const int* p_tok = (const int*)d_in[0];
  const int* h_tok = (const int*)d_in[1];
  const float* emb     = (const float*)d_in[2];
  const float* trans_w = (const float*)d_in[3];
  const float* trans_b = (const float*)d_in[4];
  const float* attn_w  = (const float*)d_in[17];
  const float* fuse_w  = (const float*)d_in[18];
  const float* fuse_b  = (const float*)d_in[19];
  const float* icnn_w  = (const float*)d_in[20];
  const float* icnn_b  = (const float*)d_in[21];
  const float* fc_w    = (const float*)d_in[49];
  const float* fc_b    = (const float*)d_in[50];

  size_t off = 0;
  auto alloc = [&](size_t bytes) -> void* {
    void* p = (char*)d_ws + off;
    off += (bytes + 255) & ~(size_t)255;
    return p;
  };

  // ---- weights ----
  __bf16* Wt_t = (__bf16*)alloc((size_t)DP * DP * 2);       // Bt 320x320
  __bf16* Whw_t[6];
  for (int i = 0; i < 6; ++i) Whw_t[i] = (__bf16*)alloc((size_t)DP * DP * 2);
  __bf16* Wfuse_t = (__bf16*)alloc((size_t)DP * 640 * 2);   // Bt 320x640
  __bf16* Aicnn = (__bf16*)alloc((size_t)96 * DP * 2);      // A_sh 96x320
  __bf16* Wd[24]; int CpkD[24];
  const int cin0[3] = {90, 125, 142};
  const int wbase[3] = {22, 31, 40};
  for (int blk = 0; blk < 3; ++blk)
    for (int l = 0; l < 8; ++l) {
      int Cin = cin0[blk] + 20 * l;
      int Cpk = rup(Cin, 32);
      CpkD[blk * 8 + l] = Cpk;
      Wd[blk * 8 + l] = (__bf16*)alloc((size_t)32 * 9 * Cpk * 2);
    }
  __bf16* Wt1 = (__bf16*)alloc((size_t)128 * 256 * 2);      // A_sh
  __bf16* Wt2 = (__bf16*)alloc((size_t)144 * 288 * 2);
  __bf16* Wt3 = (__bf16*)alloc((size_t)160 * 320 * 2);

  // ---- temps ----
  __bf16* Xg   = (__bf16*)alloc((size_t)ROWS * DP * 2);     // A_sh
  float*  xf   = (float*)alloc((size_t)ROWS * DP * 4);
  __bf16* xsh  = (__bf16*)alloc((size_t)ROWS * DP * 2);     // A_sh
  __bf16* xnat = (__bf16*)alloc((size_t)ROWS * DP * 2);     // Bt (rows l, cols d)
  float*  pg   = (float*)alloc((size_t)ROWS * DP * 4);
  float*  pn   = (float*)alloc((size_t)ROWS * DP * 4);
  float*  pl   = (float*)alloc((size_t)ROWS * DP * 4);
  float*  up   = (float*)alloc((size_t)ROWS * 4);
  float*  mid  = (float*)alloc((size_t)ROWS * 4);
  __bf16* xw   = (__bf16*)alloc((size_t)ROWS * DP * 2);     // A_sh
  __bf16* xT   = (__bf16*)alloc((size_t)BATCH * DP * SEQ * 2);  // Bt for att
  float*  S    = (float*)alloc((size_t)BATCH * SEQ * SEQ * 4);
  __bf16* Sbf  = (__bf16*)alloc((size_t)BATCH * SEQ * SEQ * 2); // A_sh
  float*  attf = (float*)alloc((size_t)ROWS * DP * 4);
  __bf16* catb = (__bf16*)alloc((size_t)ROWS * 640 * 2);    // A_sh
  float*  pref = (float*)alloc((size_t)ROWS * DP * 4);
  __bf16* encp_sh  = (__bf16*)alloc((size_t)ROWS * DP * 2);
  __bf16* encp_nat = (__bf16*)alloc((size_t)ROWS * DP * 2);
  __bf16* ench_sh  = (__bf16*)alloc((size_t)ROWS * DP * 2);
  __bf16* ench_nat = (__bf16*)alloc((size_t)ROWS * DP * 2);
  // NHWC activations
  __bf16* act1 = (__bf16*)alloc((size_t)BATCH * 9216 * 256 * 2);
  __bf16* t1o  = (__bf16*)alloc((size_t)BATCH * 9216 * 128 * 2);
  __bf16* act2 = (__bf16*)alloc((size_t)BATCH * 2304 * 288 * 2);
  __bf16* t2o  = (__bf16*)alloc((size_t)BATCH * 2304 * 144 * 2);
  __bf16* act3 = (__bf16*)alloc((size_t)BATCH * 576 * 320 * 2);
  __bf16* t3o  = (__bf16*)alloc((size_t)BATCH * 576 * 160 * 2);
  __bf16* act4 = (__bf16*)alloc((size_t)BATCH * 144 * 152 * 2);

  // ---- convert weights ----
  auto padt = [&](const float* s, __bf16* d, int sK, int sN, int dR, int dC) {
    int tot = dR * dC;
    k_padw_t<<<(tot + 255) / 256, 256, 0, stream>>>(s, d, sK, sN, dR, dC);
  };
  auto padsh = [&](const float* s, __bf16* d, int sR, int sC, int dR, int dC) {
    int tot = dR * dC;
    k_padw_sh<<<(tot + 255) / 256, 256, 0, stream>>>(s, d, sR, sC, dR, dC);
  };
  padt(trans_w, Wt_t, DIM, DIM, DP, DP);
  for (int i = 0; i < 2; ++i) {
    padt((const float*)d_in[5 + i * 6 + 0], Whw_t[i * 3 + 0], DIM, DIM, DP, DP);
    padt((const float*)d_in[5 + i * 6 + 2], Whw_t[i * 3 + 1], DIM, DIM, DP, DP);
    padt((const float*)d_in[5 + i * 6 + 4], Whw_t[i * 3 + 2], DIM, DIM, DP, DP);
  }
  padt(fuse_w, Wfuse_t, 2 * DIM, DIM, DP, 640);
  padsh(icnn_w, Aicnn, 90, DIM, 96, DP);
  for (int blk = 0; blk < 3; ++blk)
    for (int l = 0; l < 8; ++l) {
      int Cin = cin0[blk] + 20 * l;
      int Cpk = CpkD[blk * 8 + l];
      int tot = 32 * 9 * Cpk;
      k_padw_conv<<<(tot + 255) / 256, 256, 0, stream>>>(
          (const float*)d_in[wbase[blk] + l], Wd[blk * 8 + l], Cin, Cpk);
    }
  padsh((const float*)d_in[30], Wt1, 125, 250, 128, 256);
  padsh((const float*)d_in[39], Wt2, 142, 285, 144, 288);
  padsh((const float*)d_in[48], Wt3, 151, 302, 160, 320);

  // ---- zero NHWC act buffers (pad channels must read as 0) ----
  auto zero16b = [&](void* p, long long bytes) {
    long long n = bytes / 16;
    k_zero<<<(unsigned)((n + 255) / 256), 256, 0, stream>>>((uint4*)p, n);
  };
  zero16b(act1, (long long)BATCH * 9216 * 256 * 2);
  zero16b(act2, (long long)BATCH * 2304 * 288 * 2);
  zero16b(act3, (long long)BATCH * 576 * 320 * 2);

  auto gemm_f = [&](const __bf16* A, int lda, long long sA, const __bf16* Bt,
                    int ldbt, long long sB, float* C, int ldc, long long sC,
                    int M, int N, int Kp, const float* bias, int biasN, int bz) {
    int Mt = (M + 15) / 16, Nt = (N + 15) / 16;
    dim3 g((Nt + 1) / 2, (Mt + 3) / 4, bz);
    k_gemm<float, false><<<g, 128, 0, stream>>>(A, lda, sA, Bt, ldbt, sB, C,
                                                ldc, sC, Kp, Mt, bias, biasN);
  };
  auto gemm_bt = [&](const __bf16* A, int lda, const __bf16* Bt, int ldbt,
                     long long sB, __bf16* C, int ldc, long long sC, int M,
                     int N, int Kp, int bz) {
    int Mt = (M + 15) / 16, Nt = (N + 15) / 16;
    dim3 g((Nt + 1) / 2, (Mt + 3) / 4, bz);
    k_gemm<__bf16, true><<<g, 128, 0, stream>>>(A, lda, 0, Bt, ldbt, sB, C,
                                                ldc, sC, Kp, Mt, nullptr, 0);
  };

  const int EP = ROWS * DP;
  auto encode = [&](const int* tok, __bf16* enc_sh, __bf16* enc_nat) {
    k_gather<<<(EP + 255) / 256, 256, 0, stream>>>(tok, emb, Xg);
    gemm_f(Xg, DP, 0, Wt_t, DP, 0, xf, DP, 0, ROWS, DIM, DP, trans_b, DIM, 1);
    k_xcvt<<<(EP + 255) / 256, 256, 0, stream>>>(xf, xsh, xnat);
    for (int hwl = 0; hwl < 2; ++hwl) {
      gemm_f(xsh, DP, 0, Whw_t[hwl * 3 + 0], DP, 0, pg, DP, 0, ROWS, DIM, DP,
             (const float*)d_in[5 + hwl * 6 + 1], DIM, 1);
      gemm_f(xsh, DP, 0, Whw_t[hwl * 3 + 1], DP, 0, pn, DP, 0, ROWS, DIM, DP,
             (const float*)d_in[5 + hwl * 6 + 3], DIM, 1);
      gemm_f(xsh, DP, 0, Whw_t[hwl * 3 + 2], DP, 0, pl, DP, 0, ROWS, DIM, DP,
             (const float*)d_in[5 + hwl * 6 + 5], DIM, 1);
      k_highway<<<(EP + 255) / 256, 256, 0, stream>>>(pg, pn, pl, xf, xsh, xnat);
    }
    k_attn_vec<<<ROWS, 128, 0, stream>>>(xf, attn_w, up, mid, xw);
    k_transpose<<<(BATCH * DP * SEQ + 255) / 256, 256, 0, stream>>>(xf, xT);
    // S[b] = (x*w_dn) @ x^T : Bt = x itself (rows l, cols d)
    gemm_f(xw, DP, (long long)SEQ * DP, xnat, DP, (long long)SEQ * DP, S, SEQ,
           (long long)SEQ * SEQ, SEQ, SEQ, DP, nullptr, 0, BATCH);
    k_softmax<<<ROWS, 128, 0, stream>>>(S, up, mid, Sbf);
    // att[b] = softmax @ x : Bt = xT (rows d, cols j)
    gemm_f(Sbf, SEQ, (long long)SEQ * SEQ, xT, SEQ, (long long)DP * SEQ, attf,
           DP, (long long)SEQ * DP, SEQ, DIM, SEQ, nullptr, 0, BATCH);
    k_cat<<<(ROWS * 640 + 255) / 256, 256, 0, stream>>>(xf, attf, catb);
    gemm_f(catb, 640, 0, Wfuse_t, 640, 0, pref, DP, 0, ROWS, DIM, 640, fuse_b,
           DIM, 1);
    k_fuse<<<(EP + 255) / 256, 256, 0, stream>>>(pref, xf, enc_sh, enc_nat);
  };

  encode(p_tok, encp_sh, encp_nat);
  encode(h_tok, ench_sh, ench_nat);

  // interaction + icnn -> act1 NHWC channels [0,90)
  {
    dim3 g(6, 2, ROWS);
    k_interact<<<g, 128, 0, stream>>>(Aicnn, encp_sh, ench_nat, icnn_b, act1, 256);
  }

  // dense block 1: 96x96, Cp=256
  int Cin = 90;
  for (int l = 0; l < 8; ++l) {
    dim3 g(9216 / 64, 1, BATCH);
    k_dense_conv<<<g, 128, 0, stream>>>(Wd[l], 9 * CpkD[l], act1, 256, 96, 96,
                                        Cin, CpkD[l]);
    Cin += 20;
  }
  gemm_bt(Wt1, 256, act1, 256, 9216LL * 256, t1o, 128, 9216LL * 128, 125, 9216,
          256, BATCH);
  {
    int tot = BATCH * 125 * 48 * 48;
    k_maxpool<<<(tot + 255) / 256, 256, 0, stream>>>(t1o, act2, 125, 128, 96,
                                                     96, 288, tot);
  }

  // dense block 2: 48x48, Cp=288
  Cin = 125;
  for (int l = 0; l < 8; ++l) {
    dim3 g(2304 / 64, 1, BATCH);
    k_dense_conv<<<g, 128, 0, stream>>>(Wd[8 + l], 9 * CpkD[8 + l], act2, 288,
                                        48, 48, Cin, CpkD[8 + l]);
    Cin += 20;
  }
  gemm_bt(Wt2, 288, act2, 288, 2304LL * 288, t2o, 144, 2304LL * 144, 142, 2304,
          288, BATCH);
  {
    int tot = BATCH * 142 * 24 * 24;
    k_maxpool<<<(tot + 255) / 256, 256, 0, stream>>>(t2o, act3, 142, 144, 48,
                                                     48, 320, tot);
  }

  // dense block 3: 24x24, Cp=320
  Cin = 142;
  for (int l = 0; l < 8; ++l) {
    dim3 g(576 / 64, 1, BATCH);
    k_dense_conv<<<g, 128, 0, stream>>>(Wd[16 + l], 9 * CpkD[16 + l], act3, 320,
                                        24, 24, Cin, CpkD[16 + l]);
    Cin += 20;
  }
  gemm_bt(Wt3, 320, act3, 320, 576LL * 320, t3o, 160, 576LL * 160, 151, 576,
          320, BATCH);
  {
    int tot = BATCH * 151 * 12 * 12;
    k_maxpool<<<(tot + 255) / 256, 256, 0, stream>>>(t3o, act4, 151, 160, 24,
                                                     24, 152, tot);
  }

  k_fc<<<BATCH * 3, 256, 0, stream>>>(act4, fc_w, fc_b, (float*)d_out);
}
